// SingleTemplateEmbedding_18176301597151
// MI455X (gfx1250) — matrix operand order repaired and checked by reading, not compiled, not run on hardware
//
#include <hip/hip_runtime.h>

// ---------------------------------------------------------------------------
// SingleTemplateEmbedding forward for MI455X (gfx1250, wave32, WMMA).
// All GEMM-shaped compute uses v_wmma_f32_16x16x32_bf16 (bf16 in, f32 acc).
// Fragment loads are vectorized (ds_load_b128 runs); K / dual / mode are
// template parameters so inner loops are branch-free straight-line code.
// act (f32, 16MB) lives in d_out; ~50MB of bf16/f32 intermediates in d_ws.
// ---------------------------------------------------------------------------

#define NRES 256
#define NPIX (NRES * NRES)

typedef __bf16 bf16;
typedef __attribute__((ext_vector_type(16))) __bf16 v16bf;
typedef __attribute__((ext_vector_type(8)))  __bf16 v8bf;
typedef __attribute__((ext_vector_type(8)))  float  v8f;

__device__ __forceinline__ v8f wmma_bf16(v16bf a, v16bf b, v8f c) {
  // 8 args: (neg_a, A, neg_b, B, c_mod, C, reuse_a, reuse_b)
  return __builtin_amdgcn_wmma_f32_16x16x32_bf16(false, a, false, b, (short)0, c,
                                                 false, false);
}

__device__ __forceinline__ v16bf concat8(v8bf lo, v8bf hi) {
  return __builtin_shufflevector(lo, hi, 0, 1, 2, 3, 4, 5, 6, 7, 8, 9, 10, 11,
                                 12, 13, 14, 15);
}

// A-matrix (16-bit 16x32) lane K-base: lanes 0-15 hold K 0-7 & 16-23,
// lanes 16-31 hold K 8-15 & 24-31 -> two contiguous 8-elem runs at +kb, +kb+16.
__device__ __forceinline__ int kbaseA(int lane) { return (lane & 16) ? 8 : 0; }
// Contiguous K-base (k = 16h + s, unused; B loads use kbaseA): lanes 0-15 K 0-15, 16-31 K 16-31
// -> one contiguous 16-elem run at +kb2.
__device__ __forceinline__ int kbaseB(int lane) { return (lane & 16) ? 16 : 0; }

__device__ __forceinline__ float sigmoidf_(float x) {
  return 1.0f / (1.0f + __expf(-x));
}

// ---------------------------------------------------------------------------
// Rigid-frame construction per residue (rot stored already-transposed: R[k][c])
// ---------------------------------------------------------------------------
__global__ void geom_kernel(const float* __restrict__ apos,
                            const float* __restrict__ amask,
                            float* __restrict__ rot, float* __restrict__ trans,
                            float* __restrict__ bbm) {
  int r = threadIdx.x;
  if (r >= NRES) return;
  const float* n  = apos + (size_t)(r * 37 + 0) * 3;
  const float* ca = apos + (size_t)(r * 37 + 1) * 3;
  const float* cp = apos + (size_t)(r * 37 + 2) * 3;
  float cx = cp[0] - ca[0], cy = cp[1] - ca[1], cz = cp[2] - ca[2];
  float nx = n[0] - ca[0], ny = n[1] - ca[1], nz = n[2] - ca[2];
  float norm1 = sqrtf(1e-20f + cx * cx + cy * cy);
  float s1 = -cy / norm1, c1v = cx / norm1;
  float norm2 = sqrtf(1e-20f + cx * cx + cy * cy + cz * cz);
  float s2 = cz / norm2, c2v = sqrtf(cx * cx + cy * cy) / norm2;
  float c1m[3][3] = {{c1v, -s1, 0.f}, {s1, c1v, 0.f}, {0.f, 0.f, 1.f}};
  float c2m[3][3] = {{c2v, 0.f, s2}, {0.f, 1.f, 0.f}, {-s2, 0.f, c2v}};
  float cr[3][3];
  for (int i = 0; i < 3; i++)
    for (int j = 0; j < 3; j++) {
      float a = 0.f;
      for (int k = 0; k < 3; k++) a += c2m[i][k] * c1m[k][j];
      cr[i][j] = a;
    }
  float r1 = cr[1][0] * nx + cr[1][1] * ny + cr[1][2] * nz;
  float r2 = cr[2][0] * nx + cr[2][1] * ny + cr[2][2] * nz;
  float norm3 = sqrtf(1e-20f + r1 * r1 + r2 * r2);
  float sn = -r2 / norm3, cn = r1 / norm3;
  float nm[3][3] = {{1.f, 0.f, 0.f}, {0.f, cn, -sn}, {0.f, sn, cn}};
  float rm[3][3];
  for (int i = 0; i < 3; i++)
    for (int j = 0; j < 3; j++) {
      float a = 0.f;
      for (int k = 0; k < 3; k++) a += nm[i][k] * cr[k][j];
      rm[i][j] = a;
    }
  for (int k = 0; k < 3; k++)
    for (int c = 0; c < 3; c++) rot[r * 9 + k * 3 + c] = rm[c][k];  // transposed
  for (int d = 0; d < 3; d++) trans[r * 3 + d] = ca[d];
  bbm[r] = amask[r * 37 + 0] * amask[r * 37 + 1] * amask[r * 37 + 2];
}

// ---------------------------------------------------------------------------
// Per-pixel 88-dim feature + embedding matmul: act = (feat*bb2) @ emb_w + emb_b
// ---------------------------------------------------------------------------
__global__ void featurize_kernel(const float* __restrict__ pb,
                                 const float* __restrict__ pbm,
                                 const int* __restrict__ aatype,
                                 const float* __restrict__ rot,
                                 const float* __restrict__ trans,
                                 const float* __restrict__ bbm,
                                 const float* __restrict__ emb_w,
                                 const float* __restrict__ emb_b,
                                 float* __restrict__ act) {
  const int p = blockIdx.x;
  const int i = p >> 8, j = p & 255;
  const int t = threadIdx.x;  // 64 threads
  __shared__ float feat[88];
  __shared__ float s_bb2;
  if (t == 0) {
    float dx = pb[i * 3 + 0] - pb[j * 3 + 0];
    float dy = pb[i * 3 + 1] - pb[j * 3 + 1];
    float dz = pb[i * 3 + 2] - pb[j * 3 + 2];
    float d2 = dx * dx + dy * dy + dz * dz;
    for (int k = 0; k < 39; k++) {
      float lo = 3.25f + 1.25f * k;
      lo = lo * lo;
      float hi = (k < 38) ? (3.25f + 1.25f * (k + 1)) : 0.f;
      hi = (k < 38) ? hi * hi : 1e8f;
      feat[k] = (d2 > lo && d2 < hi) ? 1.f : 0.f;
    }
    feat[39] = pbm[i] * pbm[j];
    float bb2 = bbm[i] * bbm[j];
    float tx = trans[j * 3 + 0] - trans[i * 3 + 0];
    float ty = trans[j * 3 + 1] - trans[i * 3 + 1];
    float tz = trans[j * 3 + 2] - trans[i * 3 + 2];
    float av[3];
    for (int c = 0; c < 3; c++)
      av[c] = rot[i * 9 + 0 + c] * tx + rot[i * 9 + 3 + c] * ty +
              rot[i * 9 + 6 + c] * tz;
    float inv = rsqrtf(1e-6f + av[0] * av[0] + av[1] * av[1] + av[2] * av[2]);
    feat[84] = av[0] * inv * bb2;
    feat[85] = av[1] * inv * bb2;
    feat[86] = av[2] * inv * bb2;
    feat[87] = bb2;
    s_bb2 = bb2;
  }
  if (t < 22) {
    feat[40 + t] = (aatype[j] == t) ? 1.f : 0.f;
    feat[62 + t] = (aatype[i] == t) ? 1.f : 0.f;
  }
  __syncthreads();
  float acc = 0.f;
  for (int k = 0; k < 88; k++) acc += feat[k] * emb_w[k * 64 + t];
  act[(size_t)p * 64 + t] = s_bb2 * acc + emb_b[t];
}

// ---------------------------------------------------------------------------
// LayerNorm over 64 channels. Optional transposed read (for "ending" passes),
// dual-typed output (bf16 for WMMA consumption, f32 for final output).
// ---------------------------------------------------------------------------
__global__ void ln_kernel(const float* __restrict__ src,
                          const float* __restrict__ s,
                          const float* __restrict__ b, bf16* __restrict__ dstB,
                          float* __restrict__ dstF, int transposed) {
  const int p = blockIdx.x;
  const int t = threadIdx.x;  // 64 threads
  const int sp = transposed ? (((p & 255) << 8) | (p >> 8)) : p;
  float v = src[(size_t)sp * 64 + t];
  __shared__ float red[64];
  red[t] = v;
  __syncthreads();
  for (int o = 32; o; o >>= 1) {
    if (t < o) red[t] += red[t + o];
    __syncthreads();
  }
  float mu = red[0] * (1.f / 64.f);
  __syncthreads();
  float d = v - mu;
  red[t] = d * d;
  __syncthreads();
  for (int o = 32; o; o >>= 1) {
    if (t < o) red[t] += red[t + o];
    __syncthreads();
  }
  float var = red[0] * (1.f / 64.f);
  float y = d * rsqrtf(var + 1e-5f) * s[t] + b[t];
  if (dstB) dstB[(size_t)p * 64 + t] = (bf16)y;
  if (dstF) dstF[(size_t)p * 64 + t] = y;
}

// nb[h,q,k] = sum_c x[q,k,c] * f2d[c,h]  (N=4 heads -> scalar is fine)
__global__ void nb_kernel(const bf16* __restrict__ x,
                          const float* __restrict__ f2d,
                          float* __restrict__ nb) {
  int p = blockIdx.x * blockDim.x + threadIdx.x;
  if (p >= NPIX) return;
  float a0 = 0, a1 = 0, a2 = 0, a3 = 0;
  for (int c = 0; c < 64; c++) {
    float xv = (float)x[(size_t)p * 64 + c];
    a0 += xv * f2d[c * 4 + 0];
    a1 += xv * f2d[c * 4 + 1];
    a2 += xv * f2d[c * 4 + 2];
    a3 += xv * f2d[c * 4 + 3];
  }
  nb[p * 4 + 0] = a0;
  nb[p * 4 + 1] = a1;
  nb[p * 4 + 2] = a2;
  nb[p * 4 + 3] = a3;
}

__global__ void transpose_mask_kernel(const float* __restrict__ m,
                                      float* __restrict__ mt) {
  int p = blockIdx.x * blockDim.x + threadIdx.x;
  if (p < NPIX) mt[p] = m[(p & 255) * 256 + (p >> 8)];
}

// ---------------------------------------------------------------------------
// Fused (dual) GEMM over pixels: M=65536, KT/N in {64,128} (compile-time KT).
// 8 waves/WG, each wave = one 16x16 WMMA tile. A block (32xKT) is a contiguous
// global region -> b128 block copy to LDS. W staged TRANSPOSED ([n][k]) so B
// fragments are contiguous 16B runs. MODE: 0=plain,1=relu,2=sigmoid,
// 3=P*sigmoid(G) (dual accumulators). Inner loop is branch-free.
// ---------------------------------------------------------------------------
template <int KT, bool DUAL, int MODE>
__global__ __launch_bounds__(256) void gemm_kernel(
    const bf16* __restrict__ A1, const float* __restrict__ W1,
    const float* __restrict__ B1, float scale1, const bf16* __restrict__ A2,
    const float* __restrict__ W2, const float* __restrict__ B2,
    const float* __restrict__ maskRow, int N, float* __restrict__ outF,
    bf16* __restrict__ outB, int residual, int transposedStore) {
  const int m0 = blockIdx.x * 32;
  const int n0 = blockIdx.y * 64;
  const int tid = threadIdx.x;
  const int lane = tid & 31, w = tid >> 5;
  __shared__ alignas(16) bf16 sA1[32 * KT];
  __shared__ alignas(16) bf16 sA2[DUAL ? 32 * KT : 8];
  __shared__ alignas(16) bf16 sW1[64 * KT];  // transposed: [n][k]
  __shared__ alignas(16) bf16 sW2[DUAL ? 64 * KT : 8];
  // A block: rows m0..m0+31 are contiguous -> vector block copy
  {
    constexpr int chunks = (32 * KT) / 8;
    const v8bf* gA1 = (const v8bf*)(A1 + (size_t)m0 * KT);
    const v8bf* gA2 = DUAL ? (const v8bf*)(A2 + (size_t)m0 * KT) : nullptr;
    for (int c = tid; c < chunks; c += 256) {
      ((v8bf*)sA1)[c] = gA1[c];
      if (DUAL) ((v8bf*)sA2)[c] = gA2[c];
    }
  }
  for (int idx = tid; idx < KT * 64; idx += 256) {
    int kq = idx >> 6, n = idx & 63;
    sW1[n * KT + kq] = (bf16)W1[kq * N + n0 + n];
    if (DUAL) sW2[n * KT + kq] = (bf16)W2[kq * N + n0 + n];
  }
  __syncthreads();
  const int wm = (w >> 2) * 16;  // 0 or 16
  const int wn = (w & 3) * 16;   // 0,16,32,48
  const int arow = wm + (lane & 15);
  const int bcol = wn + (lane & 15);
  const int kbA = kbaseA(lane);
  const int kbB = kbaseB(lane);
  v8f acc1 = {};
  v8f acc2 = {};
#pragma unroll
  for (int kk = 0; kk < KT; kk += 32) {
    const bf16* pa1 = sA1 + arow * KT + kk + kbA;
    const bf16* pb1 = sW1 + bcol * KT + kk + kbA;
    v16bf a1 = concat8(*(const v8bf*)pa1, *(const v8bf*)(pa1 + 16));
    v16bf b1 = concat8(*(const v8bf*)pb1, *(const v8bf*)(pb1 + 16));
    acc1 = wmma_bf16(a1, b1, acc1);
    if (DUAL) {
      const bf16* pa2 = sA2 + arow * KT + kk + kbA;
      const bf16* pb2 = sW2 + bcol * KT + kk + kbA;
      v16bf a2 = concat8(*(const v8bf*)pa2, *(const v8bf*)(pa2 + 16));
      v16bf b2 = concat8(*(const v8bf*)pb2, *(const v8bf*)(pb2 + 16));
      acc2 = wmma_bf16(a2, b2, acc2);
    }
  }
  const int hi = (lane >> 4) * 8;
#pragma unroll
  for (int r = 0; r < 8; r++) {
    int gm = m0 + wm + r + hi;
    int gn = n0 + wn + (lane & 15);
    float v = acc1[r] * scale1 + (B1 ? B1[gn] : 0.f);
    if (MODE == 1)
      v = fmaxf(v, 0.f);
    else if (MODE == 2)
      v = sigmoidf_(v);
    else if (MODE == 3) {
      float g = acc2[r] + (B2 ? B2[gn] : 0.f);
      v *= sigmoidf_(g);
    }
    if (maskRow) v *= maskRow[gm];
    if (residual) {
      size_t oi = transposedStore
                      ? ((size_t)((gm & 255) * 256 + (gm >> 8))) * 64 + gn
                      : (size_t)gm * (size_t)N + gn;
      outF[oi] += v;
    } else if (outB) {
      outB[(size_t)gm * (size_t)N + gn] = (bf16)v;
    } else {
      outF[(size_t)gm * (size_t)N + gn] = v;
    }
  }
}

// ---------------------------------------------------------------------------
// Attention core: one wave per (head, q-tile, row b).
// q/k fragments are direct 16B global vector loads (d=16 padded to K=32;
// elements 8-15 of each lane's q/k fragment (K 16..31) are zeros). V is staged
// transposed in LDS once so P.V fragments are contiguous 16B LDS reads.
// ---------------------------------------------------------------------------
__global__ __launch_bounds__(32) void attn_kernel(
    const bf16* __restrict__ q, const bf16* __restrict__ k,
    const bf16* __restrict__ v, const bf16* __restrict__ gate,
    const float* __restrict__ nb, const float* __restrict__ mask,
    bf16* __restrict__ wa) {
  const int h = blockIdx.x;
  const int q0 = blockIdx.y * 16;
  const int b = blockIdx.z;
  const int lane = threadIdx.x;
  __shared__ alignas(16) float slog[16 * 256];
  __shared__ alignas(16) bf16 sprob[16 * 256];
  __shared__ alignas(16) bf16 vT[16 * 256];  // [d][key]
  const v8bf z8 = {};
  // stage V transposed: [d][key]
  for (int key = lane; key < 256; key += 32) {
    const bf16* vp = v + ((size_t)(b * 256 + key)) * 64 + h * 16;
    v8bf lo = *(const v8bf*)vp;
    v8bf hi = *(const v8bf*)(vp + 8);
#pragma unroll
    for (int dd = 0; dd < 8; dd++) vT[dd * 256 + key] = lo[dd];
#pragma unroll
    for (int dd = 0; dd < 8; dd++) vT[(8 + dd) * 256 + key] = hi[dd];
  }
  const int kbA = kbaseA(lane);
  const int kbB = kbaseB(lane);
  // q fragment: lane's real K range is 8 contiguous d values, rest is padding
  const bf16* qp =
      q + ((size_t)(b * 256 + q0 + (lane & 15))) * 64 + h * 16 + kbA;
  v16bf af = concat8(*(const v8bf*)qp, z8);
  for (int kt = 0; kt < 16; kt++) {
    int key = kt * 16 + (lane & 15);
    v16bf bfg;
    {  // lane half h holds K 8h..8h+7 real; K 16..31 = padding
      const bf16* kp = k + ((size_t)(b * 256 + key)) * 64 + h * 16;
      bfg = concat8(*(const v8bf*)(kp + kbA), z8);
    }
    v8f cc = {};
    cc = wmma_bf16(af, bfg, cc);
    int hi = (lane >> 4) * 8;
    float bias = 1e9f * (mask[b * 256 + key] - 1.f);
#pragma unroll
    for (int r = 0; r < 8; r++) {
      int qq = r + hi;
      slog[qq * 256 + key] =
          cc[r] + bias + nb[((size_t)(q0 + qq) * 256 + key) * 4 + h];
    }
  }
  __syncthreads();
  for (int r = 0; r < 16; r++) {
    float mx = -1e30f;
    for (int c = lane; c < 256; c += 32) mx = fmaxf(mx, slog[r * 256 + c]);
    for (int o = 16; o; o >>= 1) mx = fmaxf(mx, __shfl_xor(mx, o, 32));
    float sum = 0.f;
    for (int c = lane; c < 256; c += 32) {
      float e = __expf(slog[r * 256 + c] - mx);
      slog[r * 256 + c] = e;
      sum += e;
    }
    for (int o = 16; o; o >>= 1) sum += __shfl_xor(sum, o, 32);
    float inv = 1.f / sum;
    for (int c = lane; c < 256; c += 32)
      sprob[r * 256 + c] = (bf16)(slog[r * 256 + c] * inv);
  }
  __syncthreads();
  v8f acc = {};
#pragma unroll
  for (int kc = 0; kc < 256; kc += 32) {
    const bf16* pa = sprob + (lane & 15) * 256 + kc + kbA;
    const bf16* pv = vT + (lane & 15) * 256 + kc + kbA;
    v16bf ap = concat8(*(const v8bf*)pa, *(const v8bf*)(pa + 16));
    v16bf bv = concat8(*(const v8bf*)pv, *(const v8bf*)(pv + 16));
    acc = wmma_bf16(ap, bv, acc);
  }
  const int hi = (lane >> 4) * 8;
  const int d = lane & 15;
#pragma unroll
  for (int r = 0; r < 8; r++) {
    int qq = q0 + r + hi;
    size_t pix = (size_t)(b * 256 + qq);
    float g = (float)gate[pix * 64 + h * 16 + d];
    wa[pix * 64 + h * 16 + d] = (bf16)(acc[r] * g);
  }
}

// ---------------------------------------------------------------------------
// Triangle einsum: per-channel 256x256x256 matmul, batched 8 channels per WG
// (8 waves, one wave per channel). LDS tiles laid out [c][row][k] so WMMA
// fragments are contiguous 16B runs. INCOMING is a template param so the
// staging loop is branch-free.
// outgoing: out[i,j,c] = sum_k L[i,k,c]*R[j,k,c]
// incoming: out[i,j,c] = sum_k L[k,j,c]*R[k,i,c]
// ---------------------------------------------------------------------------
template <bool INCOMING>
__global__ __launch_bounds__(256) void trimul_kernel(
    const bf16* __restrict__ left, const bf16* __restrict__ right,
    float* __restrict__ tri) {
  const int i0 = blockIdx.x * 16, j0 = blockIdx.y * 16, c0 = blockIdx.z * 8;
  const int tid = threadIdx.x, lane = tid & 31, cw = tid >> 5;
  __shared__ alignas(16) bf16 sL[8 * 16 * 32];  // [c][row][k]
  __shared__ alignas(16) bf16 sR[8 * 16 * 32];
  const int kbA = kbaseA(lane);
  const int kbB = kbaseB(lane);
  v8f acc = {};
  for (int k0 = 0; k0 < 256; k0 += 32) {
    __syncthreads();
    for (int idx = tid; idx < 16 * 32 * 8; idx += 256) {
      int c = idx & 7, kk = (idx >> 3) & 31, r = idx >> 8;
      int kg = k0 + kk;
      int dst = (c * 16 + r) * 32 + kk;
      if (!INCOMING) {
        sL[dst] = left[((size_t)((i0 + r) * 256 + kg)) * 64 + c0 + c];
        sR[dst] = right[((size_t)((j0 + r) * 256 + kg)) * 64 + c0 + c];
      } else {
        sL[dst] = right[((size_t)(kg * 256 + (i0 + r))) * 64 + c0 + c];
        sR[dst] = left[((size_t)(kg * 256 + (j0 + r))) * 64 + c0 + c];
      }
    }
    if (k0 + 32 < 256) {  // hint next K-chunk into cache (global_prefetch_b8)
      int kn = k0 + 32 + ((tid >> 3) & 31);
      const bf16* pf =
          !INCOMING ? left + ((size_t)((i0 + (tid & 7)) * 256 + kn)) * 64 + c0
                    : right + ((size_t)(kn * 256 + i0 + (tid & 7))) * 64 + c0;
      __builtin_prefetch(pf, 0, 1);
    }
    __syncthreads();
    const bf16* pa = sL + (cw * 16 + (lane & 15)) * 32 + kbA;
    const bf16* pb = sR + (cw * 16 + (lane & 15)) * 32 + kbA;
    v16bf a = concat8(*(const v8bf*)pa, *(const v8bf*)(pa + 16));
    v16bf bb = concat8(*(const v8bf*)pb, *(const v8bf*)(pb + 16));
    acc = wmma_bf16(a, bb, acc);
  }
  const int hi = (lane >> 4) * 8;
#pragma unroll
  for (int r = 0; r < 8; r++) {
    tri[((size_t)((i0 + r + hi) * 256 + (j0 + (lane & 15)))) * 64 + c0 + cw] =
        acc[r];
  }
}

// ---------------------------------------------------------------------------
extern "C" void kernel_launch(void* const* d_in, const int* in_sizes, int n_in,
                              void* d_out, int out_size, void* d_ws,
                              size_t ws_size, hipStream_t stream) {
  (void)in_sizes; (void)n_in; (void)out_size; (void)ws_size;
  const int* aatype = (const int*)d_in[1];
  const float* pbm = (const float*)d_in[2];
  const float* pb = (const float*)d_in[3];
  const float* apos = (const float*)d_in[4];
  const float* amask = (const float*)d_in[5];
  const float* mask2d = (const float*)d_in[6];
  const float* emb_w = (const float*)d_in[7];
  const float* emb_b = (const float*)d_in[8];
  const float* out_ln_s = (const float*)d_in[9];
  const float* out_ln_b = (const float*)d_in[10];
  const float* ta_ln_s = (const float*)d_in[11];
  const float* ta_ln_b = (const float*)d_in[12];
  const float* ta_f2d = (const float*)d_in[13];
  const float* ta_qw = (const float*)d_in[14];
  const float* ta_kw = (const float*)d_in[15];
  const float* ta_vw = (const float*)d_in[16];
  const float* ta_gw = (const float*)d_in[17];
  const float* ta_gb = (const float*)d_in[18];
  const float* ta_ow = (const float*)d_in[19];
  const float* ta_ob = (const float*)d_in[20];
  const float* tm_ln_s = (const float*)d_in[21];
  const float* tm_ln_b = (const float*)d_in[22];
  const float* tm_lpw = (const float*)d_in[23];
  const float* tm_lpb = (const float*)d_in[24];
  const float* tm_lgw = (const float*)d_in[25];
  const float* tm_lgb = (const float*)d_in[26];
  const float* tm_rpw = (const float*)d_in[27];
  const float* tm_rpb = (const float*)d_in[28];
  const float* tm_rgw = (const float*)d_in[29];
  const float* tm_rgb = (const float*)d_in[30];
  const float* tm_cln_s = (const float*)d_in[31];
  const float* tm_cln_b = (const float*)d_in[32];
  const float* tm_ow = (const float*)d_in[33];
  const float* tm_ob = (const float*)d_in[34];
  const float* tm_gw = (const float*)d_in[35];
  const float* tm_gb = (const float*)d_in[36];
  const float* tr_ln_s = (const float*)d_in[37];
  const float* tr_ln_b = (const float*)d_in[38];
  const float* tr_w1 = (const float*)d_in[39];
  const float* tr_b1 = (const float*)d_in[40];
  const float* tr_w2 = (const float*)d_in[41];
  const float* tr_b2 = (const float*)d_in[42];

  float* act = (float*)d_out;  // 256*256*64 f32 residual stream lives in d_out
  char* ws = (char*)d_ws;
  const size_t MB = (size_t)1 << 20;
  bf16* xB = (bf16*)(ws + 0 * MB);    // LN'd activations (bf16)
  bf16* bufA = (bf16*)(ws + 8 * MB);  // q / left / transition-hidden(lo)
  bf16* bufB = (bf16*)(ws + 16 * MB); // k / right / transition-hidden(hi)
  bf16* bufC = (bf16*)(ws + 24 * MB); // v
  bf16* bufD = (bf16*)(ws + 32 * MB); // attention gate
  bf16* bufE = (bf16*)(ws + 40 * MB); // wa / tri_ln
  float* triF = (float*)(ws + 24 * MB);  // 16MB f32, spans C+D (tri_mult only)
  float* nbB = (float*)(ws + 48 * MB);
  float* maskT = (float*)(ws + 49 * MB);
  float* rotB = (float*)(ws + 50 * MB);
  float* transB = rotB + NRES * 9;
  float* bbmB = transB + NRES * 3;
  bf16* hB = bufA;  // transition hidden: 65536x128 bf16 = 16MB over A+B

  geom_kernel<<<1, 256, 0, stream>>>(apos, amask, rotB, transB, bbmB);
  transpose_mask_kernel<<<NPIX / 256, 256, 0, stream>>>(mask2d, maskT);
  featurize_kernel<<<NPIX, 64, 0, stream>>>(pb, pbm, aatype, rotB, transB,
                                            bbmB, emb_w, emb_b, act);
  const dim3 gB(256);
  const dim3 g1(NPIX / 32, 1);
  for (int blk = 0; blk < 2; blk++) {
    // ---- triangle attention: starting (pass 0) / ending (pass 1, transposed)
    for (int pass = 0; pass < 2; pass++) {
      int wi = blk * 2 + pass;
      int ending = pass;
      ln_kernel<<<NPIX, 64, 0, stream>>>(act, ta_ln_s + wi * 64,
                                         ta_ln_b + wi * 64, xB, nullptr, ending);
      nb_kernel<<<NPIX / 256, 256, 0, stream>>>(xB, ta_f2d + wi * 256, nbB);
      gemm_kernel<64, false, 0><<<g1, gB, 0, stream>>>(
          xB, ta_qw + wi * 4096, nullptr, 0.25f, nullptr, nullptr, nullptr,
          nullptr, 64, nullptr, bufA, 0, 0);
      gemm_kernel<64, false, 0><<<g1, gB, 0, stream>>>(
          xB, ta_kw + wi * 4096, nullptr, 1.f, nullptr, nullptr, nullptr,
          nullptr, 64, nullptr, bufB, 0, 0);
      gemm_kernel<64, false, 0><<<g1, gB, 0, stream>>>(
          xB, ta_vw + wi * 4096, nullptr, 1.f, nullptr, nullptr, nullptr,
          nullptr, 64, nullptr, bufC, 0, 0);
      gemm_kernel<64, false, 2><<<g1, gB, 0, stream>>>(
          xB, ta_gw + wi * 4096, ta_gb + wi * 64, 1.f, nullptr, nullptr,
          nullptr, nullptr, 64, nullptr, bufD, 0, 0);
      attn_kernel<<<dim3(4, 16, 256), 32, 0, stream>>>(
          bufA, bufB, bufC, bufD, nbB, ending ? maskT : mask2d, bufE);
      gemm_kernel<64, false, 0><<<g1, gB, 0, stream>>>(
          bufE, ta_ow + wi * 4096, ta_ob + wi * 64, 1.f, nullptr, nullptr,
          nullptr, nullptr, 64, act, nullptr, 1, ending);
    }
    // ---- triangle multiplication: outgoing (pass 0) / incoming (pass 1)
    for (int pass = 0; pass < 2; pass++) {
      int wi = blk * 2 + pass;
      ln_kernel<<<NPIX, 64, 0, stream>>>(act, tm_ln_s + wi * 64,
                                         tm_ln_b + wi * 64, xB, nullptr, 0);
      gemm_kernel<64, true, 3><<<g1, gB, 0, stream>>>(
          xB, tm_lpw + wi * 4096, tm_lpb + wi * 64, 1.f, xB,
          tm_lgw + wi * 4096, tm_lgb + wi * 64, mask2d, 64, nullptr, bufA, 0,
          0);
      gemm_kernel<64, true, 3><<<g1, gB, 0, stream>>>(
          xB, tm_rpw + wi * 4096, tm_rpb + wi * 64, 1.f, xB,
          tm_rgw + wi * 4096, tm_rgb + wi * 64, mask2d, 64, nullptr, bufB, 0,
          0);
      if (pass == 0)
        trimul_kernel<false><<<dim3(16, 16, 8), 256, 0, stream>>>(bufA, bufB,
                                                                  triF);
      else
        trimul_kernel<true><<<dim3(16, 16, 8), 256, 0, stream>>>(bufA, bufB,
                                                                 triF);
      ln_kernel<<<NPIX, 64, 0, stream>>>(triF, tm_cln_s + wi * 64,
                                         tm_cln_b + wi * 64, bufE, nullptr, 0);
      gemm_kernel<64, true, 3><<<g1, gB, 0, stream>>>(
          bufE, tm_ow + wi * 4096, tm_ob + wi * 64, 1.f, xB, tm_gw + wi * 4096,
          tm_gb + wi * 64, nullptr, 64, act, nullptr, 1, 0);
    }
    // ---- transition
    ln_kernel<<<NPIX, 64, 0, stream>>>(act, tr_ln_s + blk * 64,
                                       tr_ln_b + blk * 64, xB, nullptr, 0);
    gemm_kernel<64, false, 1><<<dim3(NPIX / 32, 2), gB, 0, stream>>>(
        xB, tr_w1 + blk * 8192, tr_b1 + blk * 128, 1.f, nullptr, nullptr,
        nullptr, nullptr, 128, nullptr, hB, 0, 0);
    gemm_kernel<128, false, 0><<<g1, gB, 0, stream>>>(
        hB, tr_w2 + blk * 8192, tr_b2 + blk * 64, 1.f, nullptr, nullptr,
        nullptr, nullptr, 64, act, nullptr, 1, 0);
  }
  // final LayerNorm in place in d_out
  ln_kernel<<<NPIX, 64, 0, stream>>>(act, out_ln_s, out_ln_b, nullptr, act, 0);
}
